// GraphAttentionLayer_58110907515027
// MI455X (gfx1250) — matrix operand order repaired and checked by reading, not compiled, not run on hardware
//
#include <hip/hip_runtime.h>
#include <hip/hip_bf16.h>

typedef __attribute__((ext_vector_type(16))) _Float16 v16h;
typedef __attribute__((ext_vector_type(2)))  __fp16   v2fp16;   // return type of cvt_pkrtz
typedef __attribute__((ext_vector_type(8)))  float    v8f;

#define NN      8192
#define FIN     128
#define FOUT    64
#define JCHUNKS 4
#define CHUNK   (NN / JCHUNKS)   // 2048 j per chunk
#define STEPS   (CHUNK / 32)     // 64 k-steps of 32
#define WPR     (NN / 32)        // 256 bitmask words per row

#define LOG2E 1.4426950408889634f

// ---------------- workspace layout (bytes) ----------------
constexpr size_t OFF_H    = 0;                                // h    f32 [8192][64]
constexpr size_t OFF_HB   = OFF_H    + (size_t)NN * FOUT * 4; // hB   f16 swizzled B layout [jb][nb][lane][8dw]
constexpr size_t OFF_S1   = OFF_HB   + (size_t)NN * FOUT * 2;
constexpr size_t OFF_S2   = OFF_S1   + (size_t)NN * 4;
constexpr size_t OFF_MAX  = OFF_S2   + (size_t)NN * 4;
constexpr size_t OFF_ADJB = OFF_MAX  + 256;                   // bitmask [8192][256] u32 = 8MB
constexpr size_t OFF_ZP   = OFF_ADJB + (size_t)NN * WPR * 4;  // Zpart [JCHUNKS][8192]
constexpr size_t OFF_OP   = OFF_ZP   + (size_t)JCHUNKS * NN * 4; // Opart [JCHUNKS][8192][64]

// ---------------- kernel 1: h = x @ W, plus f16 B-swizzled copy ----------------
__global__ __launch_bounds__(256) void gat_hW(const float* __restrict__ x,
                                              const float* __restrict__ W,
                                              float* __restrict__ h,
                                              unsigned short* __restrict__ hB) {
  const int tid = threadIdx.x;
  const int row = blockIdx.x * 4 + (tid >> 6);  // node index j
  const int o   = tid & 63;                     // output feature n
  const float* xr = x + (size_t)row * FIN;
  float acc = 0.0f;
#pragma unroll 8
  for (int k = 0; k < FIN; ++k) acc = fmaf(xr[k], W[k * FOUT + o], acc);
  h[(size_t)row * FOUT + o] = acc;

  // scatter into this code's contiguous-k B layout (same order as A in gat_attn) (16-bit B 32x16, wave32):
  // lanes 0-15 hold K=0-15 (2 per VGPR), lanes 16-31 hold K=16-31
  const int jb = row >> 5, k5 = row & 31;
  const int nb = o >> 4,   nnc = o & 15;
  const int lane = nnc + ((k5 & 16) ? 16 : 0);
  const int v    = (k5 & 15) >> 1;
  const int hsel = k5 & 1;
  const size_t idx = ((((size_t)jb * 4 + nb) * 32 + lane) * 8 + v) * 2 + hsel;
  union { _Float16 hf; unsigned short us; } cvt;
  cvt.hf = (_Float16)acc;
  hB[idx] = cvt.us;
}

// ---------------- kernel 2: s1 = h@a1, s2 = h@a2 ----------------
__global__ __launch_bounds__(256) void gat_scores(const float* __restrict__ h,
                                                  const float* __restrict__ a,
                                                  float* __restrict__ s1,
                                                  float* __restrict__ s2) {
  const int i = blockIdx.x * 256 + threadIdx.x;
  const float* hr = h + (size_t)i * FOUT;
  float r1 = 0.0f, r2 = 0.0f;
#pragma unroll 8
  for (int o = 0; o < FOUT; ++o) {
    r1 = fmaf(hr[o], a[o], r1);
    r2 = fmaf(hr[o], a[FOUT + o], r2);
  }
  s1[i] = r1;
  s2[i] = r2;
}

// ---------------- kernel 3: global max of s2 ----------------
__global__ __launch_bounds__(256) void gat_maxs2(const float* __restrict__ s2,
                                                 float* __restrict__ mx) {
  __shared__ float red[256];
  const int tid = threadIdx.x;
  float m = -3.4e38f;
  for (int i = tid; i < NN; i += 256) m = fmaxf(m, s2[i]);
  red[tid] = m;
  __syncthreads();
  for (int s = 128; s > 0; s >>= 1) {
    if (tid < s) red[tid] = fmaxf(red[tid], red[tid + s]);
    __syncthreads();
  }
  if (tid == 0) mx[0] = red[0];
}

// ---------------- kernel 4: pack adj (256MB) -> bitmask (8MB), single streaming pass ----------------
__global__ __launch_bounds__(256) void gat_pack(const int* __restrict__ adj,
                                                unsigned* __restrict__ adjbits) {
  const int i    = blockIdx.x;
  const int wave = threadIdx.x >> 5;
  const int lane = threadIdx.x & 31;
  const int* rowp = adj + (size_t)i * NN;
  for (int it = 0; it < 32; ++it) {
    const int widx = wave * 32 + it;
    if (it + 2 < 32)
      __builtin_prefetch(rowp + (widx + 2) * 32 + lane, 0, 0);   // global_prefetch_b8
    const int v = __builtin_nontemporal_load(rowp + widx * 32 + lane);
    const unsigned m = __builtin_amdgcn_ballot_w32(v > 0);
    if (lane == 0) adjbits[(size_t)i * WPR + widx] = m;
  }
}

// ---------------- kernel 5: fused masked softmax-numerator + (P @ H) via WMMA ----------------
// one wave = one 16-row block x one j-chunk; partials combined later (deterministic, no atomics)
__global__ __launch_bounds__(256) void gat_attn(const unsigned* __restrict__ adjbits,
                                                const float* __restrict__ s1,
                                                const float* __restrict__ s2,
                                                const float* __restrict__ mx,
                                                const unsigned* __restrict__ hB,
                                                float* __restrict__ Zp,
                                                float* __restrict__ Op) {
  __shared__ __align__(16) float s2l[CHUNK];   // pre-scaled by log2(e)
  const int tid  = threadIdx.x;
  const int wave = tid >> 5;
  const int lane = tid & 31;
  const int chunk = blockIdx.y;
  const int rowblock = blockIdx.x * 8 + wave;
  const int i0 = rowblock * 16;
  const int jstart = chunk * CHUNK;

  for (int idx = tid; idx < CHUNK; idx += 256) s2l[idx] = s2[jstart + idx] * LOG2E;
  __syncthreads();

  const int M  = lane & 15;   // matrix row within block (A layout); also N col for C tiles
  const int hi = lane >> 4;   // which K half this lane provides for A
  const int i  = i0 + M;
  const float s1i = s1[i];
  const float Mi  = fmaxf(s1i + mx[0], 0.0f);  // upper bound of every logit in row i
  // exp(max(s1+s2,0) - Mi) == exp2( max(s2*log2e - c1, cmin) )
  const float c1   = (Mi - s1i) * LOG2E;
  const float cmin = -Mi * LOG2E;

  v8f c0 = {}, c1v = {}, c2v = {}, c3v = {};
  float zacc = 0.0f;
  const unsigned* adjrow = adjbits + (size_t)i * WPR;

  for (int step = 0; step < STEPS; ++step) {
    const int jw = (jstart >> 5) + step;        // global 32-wide j block index
    const unsigned word = adjrow[jw];
    const float2* s2p = (const float2*)(s2l + step * 32);

    // Build A (16x32 f16 probabilities) in this code's contiguous-k order (matches hB), fully branchless:
    // lane group hi=0: K in {0..15}; hi=1: K in {16..31}; VGPR t holds (K0,K0+1)
    union { unsigned u[8]; v16h v; } A;
#pragma unroll
    for (int t = 0; t < 8; ++t) {
      const int K0 = hi * 16 + 2 * t;
      const float2 sv = s2p[K0 >> 1];                       // ds_load_b64
      const float e0 = __builtin_amdgcn_exp2f(fmaxf(sv.x - c1, cmin));
      const float e1 = __builtin_amdgcn_exp2f(fmaxf(sv.y - c1, cmin));
      const float p0 = (word & (1u << K0))       ? e0 : 0.0f;   // v_cndmask, no branch
      const float p1 = (word & (1u << (K0 + 1))) ? e1 : 0.0f;
      zacc += p0 + p1;
      union { v2fp16 h; unsigned u; } pk;
      pk.h = __builtin_amdgcn_cvt_pkrtz(p0, p1);            // v_cvt_pk_f16_f32 (RTZ)
      A.u[t] = pk.u;
    }

    // B tiles: pre-swizzled f16 H, 32 contiguous bytes per lane -> coalesced b128 loads
    const unsigned* bbase = hB + (size_t)jw * 1024;
    const v16h b0 = *(const v16h*)(bbase + (size_t)(0 * 32 + lane) * 8);
    const v16h b1 = *(const v16h*)(bbase + (size_t)(1 * 32 + lane) * 8);
    const v16h b2 = *(const v16h*)(bbase + (size_t)(2 * 32 + lane) * 8);
    const v16h b3 = *(const v16h*)(bbase + (size_t)(3 * 32 + lane) * 8);

    c0  = __builtin_amdgcn_wmma_f32_16x16x32_f16(false, A.v, false, b0, (short)0, c0,  false, false);
    c1v = __builtin_amdgcn_wmma_f32_16x16x32_f16(false, A.v, false, b1, (short)0, c1v, false, false);
    c2v = __builtin_amdgcn_wmma_f32_16x16x32_f16(false, A.v, false, b2, (short)0, c2v, false, false);
    c3v = __builtin_amdgcn_wmma_f32_16x16x32_f16(false, A.v, false, b3, (short)0, c3v, false, false);
  }

  // Z: lanes L and L^16 hold the two K-halves for the same row -> combine
  const float zo = __shfl_xor(zacc, 16, 32);
  if (hi == 0) Zp[(size_t)chunk * NN + i] = zacc + zo;

  // C/D layout: lane&15 = N, VGPR r -> M = r + 8*(lane>=16)
#pragma unroll
  for (int r = 0; r < 8; ++r) {
    const size_t row  = (size_t)i0 + r + 8 * hi;
    const size_t base = ((size_t)chunk * NN + row) * FOUT + M;
    Op[base + 0]  = c0[r];
    Op[base + 16] = c1v[r];
    Op[base + 32] = c2v[r];
    Op[base + 48] = c3v[r];
  }
}

// ---------------- kernel 6: combine chunk partials, normalize ----------------
__global__ __launch_bounds__(256) void gat_final(const float* __restrict__ Zp,
                                                 const float* __restrict__ Op,
                                                 float* __restrict__ out) {
  const size_t idx = (size_t)blockIdx.x * 256 + threadIdx.x;  // < NN*FOUT
  const size_t i = idx >> 6;
  float z = 0.0f, v = 0.0f;
#pragma unroll
  for (int c = 0; c < JCHUNKS; ++c) {
    z += Zp[(size_t)c * NN + i];
    v += Op[(size_t)c * NN * FOUT + idx];
  }
  out[idx] = v / fmaxf(z, 1e-30f);
}

extern "C" void kernel_launch(void* const* d_in, const int* in_sizes, int n_in,
                              void* d_out, int out_size, void* d_ws, size_t ws_size,
                              hipStream_t stream) {
  (void)in_sizes; (void)n_in; (void)out_size; (void)ws_size;
  const float* x   = (const float*)d_in[0];
  const int*   adj = (const int*)d_in[1];
  const float* W   = (const float*)d_in[2];
  const float* a   = (const float*)d_in[3];
  float* out = (float*)d_out;

  char* ws = (char*)d_ws;
  float*          h       = (float*)(ws + OFF_H);
  unsigned short* hB_w    = (unsigned short*)(ws + OFF_HB);
  const unsigned* hB_r    = (const unsigned*)(ws + OFF_HB);
  float*          s1      = (float*)(ws + OFF_S1);
  float*          s2      = (float*)(ws + OFF_S2);
  float*          mx      = (float*)(ws + OFF_MAX);
  unsigned*       adjbits = (unsigned*)(ws + OFF_ADJB);
  float*          Zp      = (float*)(ws + OFF_ZP);
  float*          Op      = (float*)(ws + OFF_OP);

  gat_hW    <<<NN / 4,            256, 0, stream>>>(x, W, h, hB_w);
  gat_scores<<<NN / 256,          256, 0, stream>>>(h, a, s1, s2);
  gat_maxs2 <<<1,                 256, 0, stream>>>(s2, mx);
  gat_pack  <<<NN,                256, 0, stream>>>(adj, adjbits);
  dim3 gAttn(NN / 16 / 8, JCHUNKS);  // 64 x 4 blocks, 8 waves/block -> 2048 independent waves
  gat_attn  <<<gAttn,             256, 0, stream>>>(adjbits, s1, s2, mx, hB_r, Zp, Op);
  gat_final <<<(NN * FOUT) / 256, 256, 0, stream>>>(Zp, Op, out);
}
